// HGRNBlock_644245094457
// MI455X (gfx1250) — matrix operand order repaired and checked by reading, not compiled, not run on hardware
//
#include <hip/hip_runtime.h>
#include <hip/hip_bf16.h>
#include <math.h>

// ---------------------------------------------------------------------------
// Types for CDNA5 WMMA (gfx1250, wave32)
// ---------------------------------------------------------------------------
typedef __attribute__((ext_vector_type(16))) __bf16 v16bf;
typedef __attribute__((ext_vector_type(8)))  __bf16 v8bf;
typedef __attribute__((ext_vector_type(4)))  __bf16 v4bf;
typedef __attribute__((ext_vector_type(8)))  float  v8f;

static constexpr int Bb   = 8;
static constexpr int Tt   = 4096;
static constexpr int Dd   = 512;
static constexpr int HID  = 1024;
static constexpr int D3   = 3 * Dd;      // 1536
static constexpr float EPS = 1e-5f;

#define WMMA_BF16(a, b, c) __builtin_amdgcn_wmma_f32_16x16x32_bf16( \
        false, (a), false, (b), (short)0, (c), false, false)

// ---------------------------------------------------------------------------
// K0: float -> bf16 weight conversion (grid-stride)
// ---------------------------------------------------------------------------
__global__ void f2bf_kernel(const float* __restrict__ in, __bf16* __restrict__ out, int n) {
    int i = blockIdx.x * blockDim.x + threadIdx.x;
    int stride = gridDim.x * blockDim.x;
    for (; i < n; i += stride) out[i] = (__bf16)in[i];
}

// ---------------------------------------------------------------------------
// Block reduction (NW waves of 32): sums two values across the block
// ---------------------------------------------------------------------------
template <int NW>
__device__ __forceinline__ void block_reduce2(float& a, float& b) {
    __shared__ float sA[NW];
    __shared__ float sB[NW];
#pragma unroll
    for (int o = 16; o > 0; o >>= 1) {
        a += __shfl_xor(a, o, 32);
        b += __shfl_xor(b, o, 32);
    }
    __syncthreads();                       // protect LDS reuse across calls
    if ((threadIdx.x & 31) == 0) { sA[threadIdx.x >> 5] = a; sB[threadIdx.x >> 5] = b; }
    __syncthreads();
    a = 0.f; b = 0.f;
#pragma unroll
    for (int i = 0; i < NW; ++i) { a += sA[i]; b += sB[i]; }
}

// ---------------------------------------------------------------------------
// K1: per-row  h = LN(x; g1,b1) ; hn = LN(h; cg,cb)  (h f32, hn bf16)
// one block (128 thr) per row, one float4 / thread (D = 512)
// ---------------------------------------------------------------------------
__global__ void __launch_bounds__(128) ln_front_kernel(
    const float* __restrict__ x,
    const float* __restrict__ g1, const float* __restrict__ b1,
    const float* __restrict__ cg, const float* __restrict__ cb,
    float* __restrict__ h, __bf16* __restrict__ hn)
{
    const size_t row = blockIdx.x;
    const int d = threadIdx.x * 4;
    const float4 xv = *(const float4*)(x + row * Dd + d);

    float s  = xv.x + xv.y + xv.z + xv.w;
    float ss = xv.x * xv.x + xv.y * xv.y + xv.z * xv.z + xv.w * xv.w;
    block_reduce2<4>(s, ss);
    float mean = s * (1.f / Dd);
    float inv  = rsqrtf(ss * (1.f / Dd) - mean * mean + EPS);

    const float4 gv = *(const float4*)(g1 + d);
    const float4 bv = *(const float4*)(b1 + d);
    float4 hv;
    hv.x = (xv.x - mean) * inv * gv.x + bv.x;
    hv.y = (xv.y - mean) * inv * gv.y + bv.y;
    hv.z = (xv.z - mean) * inv * gv.z + bv.z;
    hv.w = (xv.w - mean) * inv * gv.w + bv.w;
    *(float4*)(h + row * Dd + d) = hv;

    float s2  = hv.x + hv.y + hv.z + hv.w;
    float ss2 = hv.x * hv.x + hv.y * hv.y + hv.z * hv.z + hv.w * hv.w;
    block_reduce2<4>(s2, ss2);
    float m2   = s2 * (1.f / Dd);
    float inv2 = rsqrtf(ss2 * (1.f / Dd) - m2 * m2 + EPS);

    const float4 cgv = *(const float4*)(cg + d);
    const float4 cbv = *(const float4*)(cb + d);
    v4bf o;
    o[0] = (__bf16)((hv.x - m2) * inv2 * cgv.x + cbv.x);
    o[1] = (__bf16)((hv.y - m2) * inv2 * cgv.y + cbv.y);
    o[2] = (__bf16)((hv.z - m2) * inv2 * cgv.z + cbv.z);
    o[3] = (__bf16)((hv.w - m2) * inv2 * cgv.w + cbv.w);
    *(v4bf*)(hn + row * Dd + d) = o;
}

// ---------------------------------------------------------------------------
// WMMA GEMM: C[M,N] = A[M,K] * B[N,K]^T + bias[N]  (+ epilogue)
// One wave per 32(M)x64(N) tile: 2 A frags x 4 B frags = 8 wmma / K-chunk,
// software-pipelined one chunk ahead with all fragments in distinct registers
// so waits can be partial and loads overlap the WMMAs.
// EPI 0: store f32      EPI 1: exact GELU -> bf16      EPI 2: +residual, f32
// ---------------------------------------------------------------------------
template <int EPI>
__global__ void __launch_bounds__(32) gemm_wmma_kernel(
    const __bf16* __restrict__ A,    // [M,K] row major
    const __bf16* __restrict__ Bm,   // [N,K] row major
    const float*  __restrict__ bias, // [N]
    float*        __restrict__ Cf,   // f32 out (EPI 0/2)
    __bf16*       __restrict__ Cb,   // bf16 out (EPI 1)
    const float*  __restrict__ Res,  // residual [M,N] (EPI 2)
    int M, int N, int K)
{
    const int lane = threadIdx.x & 31;
    const int half = lane >> 4;      // 0: lanes 0-15, 1: lanes 16-31
    const int l16  = lane & 15;
    const int m0 = blockIdx.y * 32;
    const int n0 = blockIdx.x * 64;

    v8f acc[8];
#pragma unroll
    for (int j = 0; j < 4; ++j) {
        const float bv = bias[n0 + j * 16 + l16];
#pragma unroll
        for (int r = 0; r < 8; ++r) { acc[j][r] = bv; acc[4 + j][r] = bv; }
    }

    // A fragment per-lane base: row = m0 (+16), K groups at half*8 and 16+half*8
    const __bf16* arow0 = A + (size_t)(m0 + l16) * K + half * 8;
    const __bf16* arow1 = arow0 + (size_t)16 * K;
    // B fragment per-lane base: N col = n0 + j*16 + l16, K groups at half*8 and 16+half*8
    const __bf16* brow0 = Bm + (size_t)(n0 +  0 + l16) * K + half * 8;
    const __bf16* brow1 = Bm + (size_t)(n0 + 16 + l16) * K + half * 8;
    const __bf16* brow2 = Bm + (size_t)(n0 + 32 + l16) * K + half * 8;
    const __bf16* brow3 = Bm + (size_t)(n0 + 48 + l16) * K + half * 8;

    auto ldA = [](const __bf16* p) -> v16bf {
        v8bf lo = *(const v8bf*)p;
        v8bf hi = *(const v8bf*)(p + 16);
        return __builtin_shufflevector(lo, hi, 0,1,2,3,4,5,6,7,8,9,10,11,12,13,14,15);
    };
    auto ldB = [](const __bf16* p) -> v16bf {
        v8bf lo = *(const v8bf*)p;
        v8bf hi = *(const v8bf*)(p + 16);
        return __builtin_shufflevector(lo, hi, 0,1,2,3,4,5,6,7,8,9,10,11,12,13,14,15);
    };

    v16bf a0 = ldA(arow0), a1 = ldA(arow1);
    v16bf b0 = ldB(brow0), b1 = ldB(brow1), b2 = ldB(brow2), b3 = ldB(brow3);

    for (int kk = 32; kk < K; kk += 32) {
        // issue next chunk's loads first (distinct regs -> partial waits)
        v16bf a0n = ldA(arow0 + kk), a1n = ldA(arow1 + kk);
        v16bf b0n = ldB(brow0 + kk), b1n = ldB(brow1 + kk);
        v16bf b2n = ldB(brow2 + kk), b3n = ldB(brow3 + kk);

        acc[0] = WMMA_BF16(a0, b0, acc[0]);
        acc[1] = WMMA_BF16(a0, b1, acc[1]);
        acc[2] = WMMA_BF16(a0, b2, acc[2]);
        acc[3] = WMMA_BF16(a0, b3, acc[3]);
        acc[4] = WMMA_BF16(a1, b0, acc[4]);
        acc[5] = WMMA_BF16(a1, b1, acc[5]);
        acc[6] = WMMA_BF16(a1, b2, acc[6]);
        acc[7] = WMMA_BF16(a1, b3, acc[7]);

        a0 = a0n; a1 = a1n; b0 = b0n; b1 = b1n; b2 = b2n; b3 = b3n;
    }
    // drain last chunk
    acc[0] = WMMA_BF16(a0, b0, acc[0]);
    acc[1] = WMMA_BF16(a0, b1, acc[1]);
    acc[2] = WMMA_BF16(a0, b2, acc[2]);
    acc[3] = WMMA_BF16(a0, b3, acc[3]);
    acc[4] = WMMA_BF16(a1, b0, acc[4]);
    acc[5] = WMMA_BF16(a1, b1, acc[5]);
    acc[6] = WMMA_BF16(a1, b2, acc[6]);
    acc[7] = WMMA_BF16(a1, b3, acc[7]);

    // C/D layout: VGPR r -> row m0 + i*16 + half*8 + r, col n0 + j*16 + l16
#pragma unroll
    for (int i = 0; i < 2; ++i) {
#pragma unroll
        for (int j = 0; j < 4; ++j) {
#pragma unroll
            for (int r = 0; r < 8; ++r) {
                const size_t idx =
                    (size_t)(m0 + i * 16 + half * 8 + r) * N + (n0 + j * 16 + l16);
                float v = acc[i * 4 + j][r];
                if (EPI == 0) {
                    Cf[idx] = v;
                } else if (EPI == 1) {
                    v = 0.5f * v * (1.0f + erff(v * 0.70710678118654752f)); // exact GELU
                    Cb[idx] = (__bf16)v;
                } else {
                    Cf[idx] = v + Res[idx];
                }
            }
        }
    }
}

// ---------------------------------------------------------------------------
// K3: HGRN sequential scan.  One block per batch, one thread per channel d.
// proj = [cand | gz | gr] per row of width 3D.  Next step's operands are
// preloaded so global loads overlap the dependent tanh/sigmoid chain, and
// global_prefetch_b8 runs 8 steps ahead.
// ---------------------------------------------------------------------------
__global__ void __launch_bounds__(512) scan_kernel(
    const float* __restrict__ proj, const float* __restrict__ u,
    float* __restrict__ y)
{
    const int b = blockIdx.x;
    const int d = threadIdx.x;            // 0..511
    const float ud = u[d];
    const float* p  = proj + (size_t)b * Tt * D3;
    float*       yp = y    + (size_t)b * Tt * Dd;

    float c  = p[d];
    float gz = p[Dd + d];
    float gr = p[2 * Dd + d];
    float s = 0.f;

    for (int t = 0; t < Tt; ++t) {
        // prefetch 8 rows ahead into caches
        if (t + 8 < Tt) {
            const float* pf = p + (size_t)(t + 8) * D3;
            __builtin_prefetch(pf + d, 0, 1);
            __builtin_prefetch(pf + Dd + d, 0, 1);
            __builtin_prefetch(pf + 2 * Dd + d, 0, 1);
        }
        // preload next step's operands (independent of recurrence)
        float cn = 0.f, gzn = 0.f, grn = 0.f;
        if (t + 1 < Tt) {
            const float* nr = p + (size_t)(t + 1) * D3;
            cn  = nr[d];
            gzn = nr[Dd + d];
            grn = nr[2 * Dd + d];
        }
        const float z  = 1.f / (1.f + expf(-gz));
        const float r  = 1.f / (1.f + expf(-gr));
        const float ht = tanhf(c + r * (ud * s));
        s = z * s + (1.f - z) * ht;
        yp[(size_t)t * Dd + d] = s;
        c = cn; gz = gzn; gr = grn;
    }
}

// ---------------------------------------------------------------------------
// K4: x2 = x + y_scan + dwconv(h) ; hn2 = LN(x2; g2,b2)   (x2 f32, hn2 bf16)
// one block (128 thr) per row, one float4 / thread
// ---------------------------------------------------------------------------
__global__ void __launch_bounds__(128) fuse_mid_kernel(
    const float* __restrict__ x,  const float* __restrict__ h,
    const float* __restrict__ ys,
    const float* __restrict__ dww, const float* __restrict__ dwb,
    const float* __restrict__ g2,  const float* __restrict__ b2,
    float* __restrict__ x2, __bf16* __restrict__ hn2)
{
    const size_t row = blockIdx.x;
    const int b = (int)(row / Tt);
    const int t = (int)(row % Tt);
    const int d = threadIdx.x * 4;

    const float4 xv = *(const float4*)(x  + row * Dd + d);
    const float4 yv = *(const float4*)(ys + row * Dd + d);

    float v[4] = { xv.x + yv.x, xv.y + yv.y, xv.z + yv.z, xv.w + yv.w };
#pragma unroll
    for (int c = 0; c < 4; ++c) v[c] += dwb[d + c];
#pragma unroll
    for (int k = 0; k < 5; ++k) {
        const int tt = t + k - 2;
        if (tt >= 0 && tt < Tt) {
            const float4 hv = *(const float4*)(h + ((size_t)b * Tt + tt) * Dd + d);
            v[0] += hv.x * dww[(d + 0) * 5 + k];
            v[1] += hv.y * dww[(d + 1) * 5 + k];
            v[2] += hv.z * dww[(d + 2) * 5 + k];
            v[3] += hv.w * dww[(d + 3) * 5 + k];
        }
    }

    float s  = v[0] + v[1] + v[2] + v[3];
    float ss = v[0] * v[0] + v[1] * v[1] + v[2] * v[2] + v[3] * v[3];
    block_reduce2<4>(s, ss);
    const float mean = s * (1.f / Dd);
    const float inv  = rsqrtf(ss * (1.f / Dd) - mean * mean + EPS);

    *(float4*)(x2 + row * Dd + d) = make_float4(v[0], v[1], v[2], v[3]);

    const float4 gv = *(const float4*)(g2 + d);
    const float4 bv = *(const float4*)(b2 + d);
    v4bf o;
    o[0] = (__bf16)((v[0] - mean) * inv * gv.x + bv.x);
    o[1] = (__bf16)((v[1] - mean) * inv * gv.y + bv.y);
    o[2] = (__bf16)((v[2] - mean) * inv * gv.z + bv.z);
    o[3] = (__bf16)((v[3] - mean) * inv * gv.w + bv.w);
    *(v4bf*)(hn2 + row * Dd + d) = o;
}

// ---------------------------------------------------------------------------
// Launch
// ---------------------------------------------------------------------------
extern "C" void kernel_launch(void* const* d_in, const int* in_sizes, int n_in,
                              void* d_out, int out_size, void* d_ws, size_t ws_size,
                              hipStream_t stream) {
    (void)in_sizes; (void)n_in; (void)out_size; (void)ws_size;

    const float* x       = (const float*)d_in[0];
    const float* norm1_g = (const float*)d_in[1];
    const float* norm1_b = (const float*)d_in[2];
    const float* u       = (const float*)d_in[3];
    const float* Wi      = (const float*)d_in[4];   // [3D, D]
    const float* bi      = (const float*)d_in[5];
    const float* cn_g    = (const float*)d_in[6];
    const float* cn_b    = (const float*)d_in[7];
    const float* dw_w    = (const float*)d_in[8];   // [D,1,5]
    const float* dw_b    = (const float*)d_in[9];
    const float* norm2_g = (const float*)d_in[10];
    const float* norm2_b = (const float*)d_in[11];
    const float* W1      = (const float*)d_in[12];  // [HID, D]
    const float* b1      = (const float*)d_in[13];
    const float* W2      = (const float*)d_in[14];  // [D, HID]
    const float* b2      = (const float*)d_in[15];
    float* out = (float*)d_out;

    const size_t M  = (size_t)Bb * Tt;              // 32768
    const size_t MB = (size_t)1 << 20;
    char* ws = (char*)d_ws;

    // Workspace layout (regions recycled once their producers/consumers retire):
    float*  proj = (float*)ws;                      // [0,192MB)   live K2..K3
    float*  x2   = (float*)ws;                      // [0,64MB)    live K4..K6 (after proj dead)
    __bf16* hn2  = (__bf16*)(ws + 64 * MB);         // [64,96MB)   live K4..K5
    __bf16* mid  = (__bf16*)(ws + 96 * MB);         // [96,160MB)  live K5..K6
    float*  h    = (float*)(ws + 192 * MB);         // [192,256MB) live K1..K4
    __bf16* hn   = (__bf16*)(ws + 256 * MB);        // [256,288MB) live K1..K2
    float*  ysc  = (float*)(ws + 256 * MB);         // [256,320MB) live K3..K4 (after hn dead)
    __bf16* Wi_b = (__bf16*)(ws + 320 * MB);        // 1.5MB
    __bf16* W1_b = (__bf16*)(ws + 322 * MB);        // 1MB
    __bf16* W2_b = (__bf16*)(ws + 324 * MB);        // 1MB

    // K0: weight conversion to bf16
    f2bf_kernel<<<1024, 256, 0, stream>>>(Wi, Wi_b, D3 * Dd);
    f2bf_kernel<<<1024, 256, 0, stream>>>(W1, W1_b, HID * Dd);
    f2bf_kernel<<<1024, 256, 0, stream>>>(W2, W2_b, Dd * HID);

    // K1: h = LN1(x), hn = LN_cell(h)
    ln_front_kernel<<<(int)M, 128, 0, stream>>>(x, norm1_g, norm1_b, cn_g, cn_b, h, hn);

    // K2: proj[M,3D] = hn @ Wi^T + bi   (v_wmma bf16)
    gemm_wmma_kernel<0><<<dim3(D3 / 64, (int)(M / 32)), 32, 0, stream>>>(
        hn, Wi_b, bi, proj, nullptr, nullptr, (int)M, D3, Dd);

    // K3: sequential HGRN scan -> ysc[M,D]
    scan_kernel<<<Bb, Dd, 0, stream>>>(proj, u, ysc);

    // K4: x2 = x + ysc + dwconv(h);  hn2 = LN2(x2)
    fuse_mid_kernel<<<(int)M, 128, 0, stream>>>(x, h, ysc, dw_w, dw_b,
                                                norm2_g, norm2_b, x2, hn2);

    // K5: mid[M,HID] = GELU(hn2 @ W1^T + b1) -> bf16   (v_wmma bf16)
    gemm_wmma_kernel<1><<<dim3(HID / 64, (int)(M / 32)), 32, 0, stream>>>(
        hn2, W1_b, b1, nullptr, mid, nullptr, (int)M, HID, Dd);

    // K6: out[M,D] = x2 + mid @ W2^T + b2   (v_wmma bf16)
    gemm_wmma_kernel<2><<<dim3(Dd / 64, (int)(M / 32)), 32, 0, stream>>>(
        mid, W2_b, b2, out, nullptr, x2, (int)M, Dd, HID);
}
